// NeRFRenderer_41463614275967
// MI455X (gfx1250) — matrix operand order repaired and checked by reading, not compiled, not run on hardware
//
#include <hip/hip_runtime.h>
#include <hip/hip_bf16.h>

typedef __attribute__((ext_vector_type(16))) _Float16 v16h;
typedef __attribute__((ext_vector_type(8)))  _Float16 v8h;
typedef __attribute__((ext_vector_type(4)))  _Float16 v4h;
typedef __attribute__((ext_vector_type(8)))  float    v8f;

#define WAVES_PER_BLOCK 4
#define BLOCK_THREADS   128
#define TC 128   // coarse steps
#define TF 128   // fine (upsample) steps
#define TA 256   // merged

struct WaveScratch {
  float zc[TC];        // coarse z
  float znew[TF];      // fine z
  float zall[TA];      // merged (sorted) z
  float sigc[TC];      // coarse sigma
  float sigall[TA];    // merged sigma
  float wc[TC];        // coarse weights
  float cdf[TC];       // 127 entries used
  float zmid[TC];      // 127 entries used
  float rgb[TA * 3];   // merged rgb
  alignas(16) _Float16 hbuf[16 * 64];  // hidden activations, permuted cols: p(h)=4*(h&15)+(h>>4)
  alignas(16) _Float16 abuf[16 * 32];  // color A input: [dir(3) | geo(15) | 0...]
};

// Wave-private LDS ordering point: LDS ops of a wave are processed in order
// (ISA: DScnt "LDS: in order"), so draining DScnt + a compiler memory barrier
// is sufficient for cross-lane RAW through this wave's own LDS scratch.
// Much cheaper than s_barrier and keeps the 4 waves of the block decoupled.
__device__ __forceinline__ void wave_lds_sync() {
  asm volatile("s_wait_dscnt 0" ::: "memory");
}

__device__ __forceinline__ float sp_softplus(float x) {
  return (x > 20.f) ? x : log1pf(__expf(x));
}
__device__ __forceinline__ float sp_sigmoid(float x) {
  return 1.f / (1.f + __expf(-x));
}
__device__ __forceinline__ float clamp1(float x) {
  return fminf(fmaxf(x, -1.f), 1.f);
}

// B fragment (32x16 f16) of row-major W[kRows][ld] at (kBase, colBase).
// Contiguous k order (as load_A), not the ISA table order: lane 0-15 -> col=lane, K=kBase+0..15 (2/VGPR); lane 16-31 -> col-16, K=+16..31.
__device__ __forceinline__ v16h load_B(const _Float16* W, int ld, int colBase,
                                       int kBase, int kRows) {
  const int lane = threadIdx.x & 31;
  const int col  = colBase + (lane & 15);
  const int koff = kBase + ((lane & 16) ? 16 : 0);
  v16h b;
#pragma unroll
  for (int e = 0; e < 16; ++e) {
    const int k = koff + e;
    b[e] = (k < kRows) ? W[k * ld + col] : (_Float16)0.f;
  }
  return b;
}

// A fragment (16x32 f16) of row-major Abuf[16][ld] from column kBase.
// Per-lane data = two contiguous 8xf16 runs -> two 16B LDS loads (ds_load_b128).
__device__ __forceinline__ v16h load_A(const _Float16* Abuf, int ld, int kBase) {
  const int lane = threadIdx.x & 31;
  const int m  = lane & 15;
  const int kh = (lane & 16) ? 16 : 0;
  const v8h lo = *(const v8h*)(Abuf + m * ld + kBase + kh);
  const v8h hh = *(const v8h*)(Abuf + m * ld + kBase + 8 + kh);
  v16h a;
#pragma unroll
  for (int e = 0; e < 8; ++e) { a[e] = lo[e]; a[8 + e] = hh[e]; }
  return a;
}

// Packed activation store: hidden units {0,1,2,3}*16+n -> physical cols 4n..4n+3 (ds_store_b64)
__device__ __forceinline__ void store_h4(_Float16* hrow, int n,
                                         float f0, float f1, float f2, float f3) {
  v4h h4;
  h4[0] = (_Float16)f0; h4[1] = (_Float16)f1; h4[2] = (_Float16)f2; h4[3] = (_Float16)f3;
  *(v4h*)(hrow + 4 * n) = h4;
}

#define WMMA(a, b, c) \
  __builtin_amdgcn_wmma_f32_16x16x32_f16(false, (a), false, (b), (short)0, (c), false, false)

__global__ __launch_bounds__(BLOCK_THREADS)
void nerf_render_fused_kernel(const float* __restrict__ rays_o,
                              const float* __restrict__ rays_d,
                              const float* __restrict__ gWd1, const float* __restrict__ gbd1,
                              const float* __restrict__ gWd2, const float* __restrict__ gbd2,
                              const float* __restrict__ gWc1, const float* __restrict__ gbc1,
                              const float* __restrict__ gWc2, const float* __restrict__ gbc2,
                              float* __restrict__ out, int N) {
  __shared__ _Float16 sWd1[3 * 64];    // density layer 1 (3 -> 64)
  __shared__ _Float16 sWd2[64 * 16];   // density layer 2, rows permuted by p(h)
  __shared__ _Float16 sWc1[18 * 64];   // color layer 1 (18 -> 64)
  __shared__ _Float16 sWc2[64 * 16];   // color layer 2 (64->3, padded to 16), rows permuted
  __shared__ float sbd1[64], sbd2[16], sbc1[64], sbc2[16];
  __shared__ WaveScratch sw[WAVES_PER_BLOCK];

  const int tid  = threadIdx.x;
  const int wave = tid >> 5;
  const int lane = tid & 31;
  const int hi   = (lane & 16) ? 1 : 0;   // C/D: M = r + 8*hi
  const int n    = lane & 15;             // C/D column within a 16-wide tile

  // ---- stage weights to LDS as f16; permute rows of layer-2 weights ----
  for (int i = tid; i < 3 * 64; i += BLOCK_THREADS)  sWd1[i] = (_Float16)gWd1[i];
  for (int i = tid; i < 18 * 64; i += BLOCK_THREADS) sWc1[i] = (_Float16)gWc1[i];
  for (int i = tid; i < 64 * 16; i += BLOCK_THREADS) {
    const int h = i >> 4, c = i & 15;
    const int p = 4 * (h & 15) + (h >> 4);        // matches store_h4 layout
    sWd2[p * 16 + c] = (_Float16)gWd2[i];
    sWc2[p * 16 + c] = (c < 3) ? (_Float16)gWc2[h * 3 + c] : (_Float16)0.f;
  }
  if (tid < 64) { sbd1[tid] = gbd1[tid]; sbc1[tid] = gbc1[tid]; }
  if (tid < 16) { sbd2[tid] = gbd2[tid]; sbc2[tid] = (tid < 3) ? gbc2[tid] : 0.f; }

  WaveScratch& S = sw[wave];
  const int ray = blockIdx.x * WAVES_PER_BLOCK + wave;

  // ---- ray constants (uniform across the wave) ----
  const float ox = rays_o[ray * 3 + 0], oy = rays_o[ray * 3 + 1], oz = rays_o[ray * 3 + 2];
  const float dx = rays_d[ray * 3 + 0], dy = rays_d[ray * 3 + 1], dz = rays_d[ray * 3 + 2];

  // ---- near/far slab test vs [-1,1]^3 ----
  const float ix = 1.f / dx, iy = 1.f / dy, iz = 1.f / dz;
  const float t1x = (-1.f - ox) * ix, t2x = (1.f - ox) * ix;
  const float t1y = (-1.f - oy) * iy, t2y = (1.f - oy) * iy;
  const float t1z = (-1.f - oz) * iz, t2z = (1.f - oz) * iz;
  const float tmin = fmaxf(fmaxf(fminf(t1x, t2x), fminf(t1y, t2y)), fminf(t1z, t2z));
  const float tmax = fminf(fminf(fmaxf(t1x, t2x), fmaxf(t1y, t2y)), fmaxf(t1z, t2z));
  const float nearv = fmaxf(tmin, 0.2f);
  const float farv  = fmaxf(tmax, nearv + 1e-4f);
  const float sdist = (farv - nearv) / (float)TC;

  __syncthreads();  // ONLY cross-wave dependence: weights staged in LDS

  // ---- hoist all loop-invariant B fragments into registers (96 VGPRs) ----
  const v16h Bd1_0 = load_B(sWd1, 64, 0,  0, 3);
  const v16h Bd1_1 = load_B(sWd1, 64, 16, 0, 3);
  const v16h Bd1_2 = load_B(sWd1, 64, 32, 0, 3);
  const v16h Bd1_3 = load_B(sWd1, 64, 48, 0, 3);
  const v16h Bd2_0 = load_B(sWd2, 16, 0, 0,  64);
  const v16h Bd2_1 = load_B(sWd2, 16, 0, 32, 64);
  const v16h Bc1_0 = load_B(sWc1, 64, 0,  0, 18);
  const v16h Bc1_1 = load_B(sWc1, 64, 16, 0, 18);
  const v16h Bc1_2 = load_B(sWc1, 64, 32, 0, 18);
  const v16h Bc1_3 = load_B(sWc1, 64, 48, 0, 18);
  const v16h Bc2_0 = load_B(sWc2, 16, 0, 0,  64);
  const v16h Bc2_1 = load_B(sWc2, 16, 0, 32, 64);

  // ---- coarse z (linspace) + zero color A-buffer (padding K=18..31 stays 0) ----
  for (int i = lane; i < TC; i += 32)
    S.zc[i] = nearv + (farv - nearv) * ((float)i / (float)(TC - 1));
  {
    v8h z8;
#pragma unroll
    for (int e = 0; e < 8; ++e) z8[e] = (_Float16)0.f;
    for (int i = lane; i < (16 * 32) / 8; i += 32) ((v8h*)S.abuf)[i] = z8;
  }
  wave_lds_sync();

  // =======================================================================
  // Phase 1: coarse density (8 tiles of 16 samples) -> sigc
  // =======================================================================
  for (int tile = 0; tile < TC / 16; ++tile) {
    const float z = S.zc[tile * 16 + n];
    const float px = clamp1(ox + dx * z);
    const float py = clamp1(oy + dy * z);
    const float pz = clamp1(oz + dz * z);
    v16h a;
#pragma unroll
    for (int e = 0; e < 16; ++e) a[e] = (_Float16)0.f;
    if (lane < 16) { a[0] = (_Float16)px; a[1] = (_Float16)py; a[2] = (_Float16)pz; }

    v8f c0 = {}, c1 = {}, c2 = {}, c3 = {};
    c0 = WMMA(a, Bd1_0, c0);
    c1 = WMMA(a, Bd1_1, c1);
    c2 = WMMA(a, Bd1_2, c2);
    c3 = WMMA(a, Bd1_3, c3);
#pragma unroll
    for (int r = 0; r < 8; ++r) {
      const int M = r + 8 * hi;
      store_h4(&S.hbuf[M * 64], n,
               fmaxf(c0[r] + sbd1[0  + n], 0.f), fmaxf(c1[r] + sbd1[16 + n], 0.f),
               fmaxf(c2[r] + sbd1[32 + n], 0.f), fmaxf(c3[r] + sbd1[48 + n], 0.f));
    }
    wave_lds_sync();
    v8f acc = {};
    acc = WMMA(load_A(S.hbuf, 64, 0),  Bd2_0, acc);
    acc = WMMA(load_A(S.hbuf, 64, 32), Bd2_1, acc);
#pragma unroll
    for (int r = 0; r < 8; ++r) {
      const int M = r + 8 * hi;
      const float v = acc[r] + sbd2[n];
      if (n == 0) S.sigc[tile * 16 + M] = sp_softplus(v);
    }
    wave_lds_sync();
  }

  // ---- coarse alpha weights + z_mid + CDF (serial, lane 0; keep rolled) ----
  if (lane == 0) {
    float Tcum = 1.f;
#pragma unroll 1
    for (int i = 0; i < TC; ++i) {
      const float delta = (i < TC - 1) ? (S.zc[i + 1] - S.zc[i]) : sdist;
      const float alpha = 1.f - __expf(-delta * S.sigc[i]);
      S.wc[i] = alpha * Tcum;
      Tcum *= (1.f - alpha + 1e-15f);
    }
#pragma unroll 1
    for (int i = 0; i < TC - 1; ++i) S.zmid[i] = 0.5f * (S.zc[i] + S.zc[i + 1]);
    float sum = 0.f;
#pragma unroll 1
    for (int j = 0; j < TC - 2; ++j) sum += S.wc[1 + j] + 1e-5f;
    float accw = 0.f;
    S.cdf[0] = 0.f;
#pragma unroll 1
    for (int j = 0; j < TC - 2; ++j) {  // cdf[0..126]
      accw += (S.wc[1 + j] + 1e-5f) / sum;
      S.cdf[j + 1] = accw;
    }
  }
  wave_lds_sync();

  // ---- inverse-CDF importance sampling (lane-parallel) ----
  {
    const float ulo = 0.5f / (float)TF;
    const float uhi = 1.f - 0.5f / (float)TF;
#pragma unroll 1
    for (int j = lane; j < TF; j += 32) {
      const float u = ulo + (uhi - ulo) * ((float)j / (float)(TF - 1));
      int l = 0, r = TC - 1;                       // 127 cdf entries
#pragma unroll 1
      while (l < r) { const int mid = (l + r) >> 1; if (S.cdf[mid] <= u) l = mid + 1; else r = mid; }
      int below = l - 1; below = below < 0 ? 0 : below;
      int above = l > (TC - 2) ? (TC - 2) : l;     // clip to 126
      const float c0v = S.cdf[below],  c1v = S.cdf[above];
      const float b0v = S.zmid[below], b1v = S.zmid[above];
      float den = c1v - c0v; if (den < 1e-5f) den = 1.f;
      S.znew[j] = b0v + ((u - c0v) / den) * (b1v - b0v);
    }
  }
  wave_lds_sync();

  // ---- rank-merge of two sorted lists (stable: coarse wins ties) ----
#pragma unroll 1
  for (int i = lane; i < TC; i += 32) {
    const float v = S.zc[i];
    int l = 0, r = TF;
#pragma unroll 1
    while (l < r) { const int mid = (l + r) >> 1; if (S.znew[mid] < v) l = mid + 1; else r = mid; }
    S.zall[i + l] = v;
  }
#pragma unroll 1
  for (int j = lane; j < TF; j += 32) {
    const float v = S.znew[j];
    int l = 0, r = TC;
#pragma unroll 1
    while (l < r) { const int mid = (l + r) >> 1; if (S.zc[mid] <= v) l = mid + 1; else r = mid; }
    S.zall[j + l] = v;
  }
  if (lane < 16) {  // direction is constant per ray: fill color-A rows once
    S.abuf[lane * 32 + 0] = (_Float16)dx;
    S.abuf[lane * 32 + 1] = (_Float16)dy;
    S.abuf[lane * 32 + 2] = (_Float16)dz;
  }
  wave_lds_sync();

  // =======================================================================
  // Phase 2: merged samples (16 tiles): density -> sigma+geo -> color -> rgb
  // =======================================================================
  for (int tile = 0; tile < TA / 16; ++tile) {
    const float z = S.zall[tile * 16 + n];
    const float px = clamp1(ox + dx * z);
    const float py = clamp1(oy + dy * z);
    const float pz = clamp1(oz + dz * z);
    v16h a;
#pragma unroll
    for (int e = 0; e < 16; ++e) a[e] = (_Float16)0.f;
    if (lane < 16) { a[0] = (_Float16)px; a[1] = (_Float16)py; a[2] = (_Float16)pz; }

    // density layer 1
    v8f c0 = {}, c1 = {}, c2 = {}, c3 = {};
    c0 = WMMA(a, Bd1_0, c0);
    c1 = WMMA(a, Bd1_1, c1);
    c2 = WMMA(a, Bd1_2, c2);
    c3 = WMMA(a, Bd1_3, c3);
#pragma unroll
    for (int r = 0; r < 8; ++r) {
      const int M = r + 8 * hi;
      store_h4(&S.hbuf[M * 64], n,
               fmaxf(c0[r] + sbd1[0  + n], 0.f), fmaxf(c1[r] + sbd1[16 + n], 0.f),
               fmaxf(c2[r] + sbd1[32 + n], 0.f), fmaxf(c3[r] + sbd1[48 + n], 0.f));
    }
    wave_lds_sync();
    // density layer 2: N=0 -> sigma (softplus); N=1..15 -> geo into color A-buffer
    v8f acc = {};
    acc = WMMA(load_A(S.hbuf, 64, 0),  Bd2_0, acc);
    acc = WMMA(load_A(S.hbuf, 64, 32), Bd2_1, acc);
#pragma unroll
    for (int r = 0; r < 8; ++r) {
      const int M = r + 8 * hi;
      const float v = acc[r] + sbd2[n];
      if (n == 0) S.sigall[tile * 16 + M] = sp_softplus(v);
      else        S.abuf[M * 32 + 2 + n] = (_Float16)v;  // geo k = 3 + (n-1)
    }
    wave_lds_sync();
    // color layer 1: A = [dir | geo], K=18 (padded to 32)
    const v16h ac = load_A(S.abuf, 32, 0);
    v8f e0 = {}, e1 = {}, e2 = {}, e3 = {};
    e0 = WMMA(ac, Bc1_0, e0);
    e1 = WMMA(ac, Bc1_1, e1);
    e2 = WMMA(ac, Bc1_2, e2);
    e3 = WMMA(ac, Bc1_3, e3);
#pragma unroll
    for (int r = 0; r < 8; ++r) {
      const int M = r + 8 * hi;
      store_h4(&S.hbuf[M * 64], n,
               fmaxf(e0[r] + sbc1[0  + n], 0.f), fmaxf(e1[r] + sbc1[16 + n], 0.f),
               fmaxf(e2[r] + sbc1[32 + n], 0.f), fmaxf(e3[r] + sbc1[48 + n], 0.f));
    }
    wave_lds_sync();
    // color layer 2 (Wc2 zero-padded to 64x16, rows permuted); sigmoid on N<3
    v8f f = {};
    f = WMMA(load_A(S.hbuf, 64, 0),  Bc2_0, f);
    f = WMMA(load_A(S.hbuf, 64, 32), Bc2_1, f);
#pragma unroll
    for (int r = 0; r < 8; ++r) {
      const int M = r + 8 * hi;
      const float v = f[r] + sbc2[n];
      if (n < 3) S.rgb[(tile * 16 + M) * 3 + n] = sp_sigmoid(v);
    }
    wave_lds_sync();
  }

  // ---- final compositing (serial scan, lane 0 of each wave; keep rolled) ----
  if (lane == 0) {
    float Tcum = 1.f, depth = 0.f, cr = 0.f, cg = 0.f, cb = 0.f, wsum = 0.f;
#pragma unroll 1
    for (int i = 0; i < TA; ++i) {
      const float delta = (i < TA - 1) ? (S.zall[i + 1] - S.zall[i]) : sdist;
      const float alpha = 1.f - __expf(-delta * S.sigall[i]);
      const float w = alpha * Tcum;
      Tcum *= (1.f - alpha + 1e-15f);
      float ozn = (S.zall[i] - nearv) / (farv - nearv);
      ozn = fminf(fmaxf(ozn, 0.f), 1.f);
      depth += w * ozn;
      cr += w * S.rgb[i * 3 + 0];
      cg += w * S.rgb[i * 3 + 1];
      cb += w * S.rgb[i * 3 + 2];
      wsum += w;
    }
    out[ray]             = depth;                  // depth  [N]
    out[N + ray * 3 + 0] = cr + (1.f - wsum);      // image  [N,3] (white bg)
    out[N + ray * 3 + 1] = cg + (1.f - wsum);
    out[N + ray * 3 + 2] = cb + (1.f - wsum);
    out[4 * N + ray]     = wsum;                   // weights_sum [N]
  }
}

extern "C" void kernel_launch(void* const* d_in, const int* in_sizes, int n_in,
                              void* d_out, int out_size, void* d_ws, size_t ws_size,
                              hipStream_t stream) {
  (void)n_in; (void)out_size; (void)d_ws; (void)ws_size;
  const float* rays_o = (const float*)d_in[0];
  const float* rays_d = (const float*)d_in[1];
  const float* Wd1 = (const float*)d_in[2];
  const float* bd1 = (const float*)d_in[3];
  const float* Wd2 = (const float*)d_in[4];
  const float* bd2 = (const float*)d_in[5];
  const float* Wc1 = (const float*)d_in[6];
  const float* bc1 = (const float*)d_in[7];
  const float* Wc2 = (const float*)d_in[8];
  const float* bc2 = (const float*)d_in[9];
  const int N = in_sizes[0] / 3;             // 32768 rays
  const int blocks = N / WAVES_PER_BLOCK;    // one ray per wave32
  hipLaunchKernelGGL(nerf_render_fused_kernel, dim3(blocks), dim3(BLOCK_THREADS), 0, stream,
                     rays_o, rays_d, Wd1, bd1, Wd2, bd2, Wc1, bc1, Wc2, bc2,
                     (float*)d_out, N);
}
